// DeepseekMoE_PT_23347442221518
// MI455X (gfx1250) — matrix operand order repaired and checked by reading, not compiled, not run on hardware
//
#include <hip/hip_runtime.h>

typedef __attribute__((ext_vector_type(16))) _Float16 v16h;
typedef __attribute__((ext_vector_type(8)))  _Float16 v8h;
typedef __attribute__((ext_vector_type(8)))  float    v8f;

namespace {
constexpr int H    = 2048;   // hidden
constexpr int M    = 1408;   // moe intermediate
constexpr int E    = 32;     // routed experts
constexpr int TOPK = 6;
constexpr int MSH  = 2816;   // shared intermediate (M * 2)
constexpr int T    = 1024;   // tokens (B*S)
constexpr int NSLOT = T * TOPK;   // 6144 routed token-slots
}

// ---------------------------------------------------------------- math helpers
__device__ __forceinline__ float fast_sigmoid(float x) {
  return __builtin_amdgcn_rcpf(1.f + __expf(-x));   // v_rcp_f32, no IEEE divide
}
__device__ __forceinline__ float silu_mul(float g, float u) {
  return g * __builtin_amdgcn_rcpf(1.f + __expf(-g)) * u;
}

// ---------------------------------------------------------------- WMMA helpers
__device__ __forceinline__ v8f wmma16(v16h a, v16h b, v8f c) {
  // D = A(16x32 f16) * B(32x16 f16) + C(16x16 f32)
  return __builtin_amdgcn_wmma_f32_16x16x32_f16(false, a, false, b, (short)0, c,
                                                false, false);
}

// A fragment: lane holds row = lane%16; j0 = k0 + 8*(lane/16) (multiple of 8).
// a[0..7] = K j0+0..7 ; a[8..15] = K j0+16..23   (16-bit A 16x32 VGPR map)
__device__ __forceinline__ v16h load_a_f32(const float* __restrict__ rowp, int j0) {
  const float4* p0 = (const float4*)(rowp + j0);        // 32B aligned
  const float4* p1 = (const float4*)(rowp + j0 + 16);
  float4 c0 = p0[0], c1 = p0[1], c2 = p1[0], c3 = p1[1];
  v16h a;
  a[0]  = (_Float16)c0.x; a[1]  = (_Float16)c0.y;
  a[2]  = (_Float16)c0.z; a[3]  = (_Float16)c0.w;
  a[4]  = (_Float16)c1.x; a[5]  = (_Float16)c1.y;
  a[6]  = (_Float16)c1.z; a[7]  = (_Float16)c1.w;
  a[8]  = (_Float16)c2.x; a[9]  = (_Float16)c2.y;
  a[10] = (_Float16)c2.z; a[11] = (_Float16)c2.w;
  a[12] = (_Float16)c3.x; a[13] = (_Float16)c3.y;
  a[14] = (_Float16)c3.z; a[15] = (_Float16)c3.w;
  return a;
}
__device__ __forceinline__ v16h load_a_f16(const _Float16* __restrict__ rowp, int j0) {
  v8h lo = *(const v8h*)(rowp + j0);        // 16B aligned (rows are 16B multiples)
  v8h hi = *(const v8h*)(rowp + j0 + 16);
  v16h a;
#pragma unroll
  for (int i = 0; i < 8; ++i) { a[i] = lo[i]; a[8 + i] = hi[i]; }
  return a;
}

// B fragment: lane holds col = lane%16; kb = k0 + 8*(lane/16); b[0..7] = K kb+0..7 ; b[8..15] = K kb+16..23.
__device__ __forceinline__ v16h load_b_f32(const float* __restrict__ w, int ld,
                                           int kb, int col) {
  v16h b;
#pragma unroll
  for (int i = 0; i < 16; ++i) b[i] = (_Float16)w[(size_t)(kb + i + (i >= 8 ? 8 : 0)) * ld + col];
  return b;
}

// ---------------------------------------------------------------- router
// one wave per token, lane == expert (E == 32 == wave32)
__global__ __launch_bounds__(32) void router_kernel(
    const float* __restrict__ x, const float* __restrict__ w_router,
    const float* __restrict__ corr_bias, int* __restrict__ expert_count,
    int* __restrict__ topk_e, float* __restrict__ topk_w, int* __restrict__ pos_arr)
{
  const int t = blockIdx.x;
  const int lane = threadIdx.x;

  float acc = 0.f;
  for (int h = 0; h < H; h += 32) {
    float xv = x[(size_t)t * H + h + lane];
#pragma unroll
    for (int i = 0; i < 32; ++i) {
      float xi = __shfl(xv, i, 32);
      acc += xi * w_router[(size_t)(h + i) * E + lane];
    }
  }
  const float score = fast_sigmoid(acc);
  const float sfc   = score + corr_bias[lane];

  // group score = sum of top-2 within the 4 experts of this group
  const int g = lane >> 2;
  float v0 = __shfl(sfc, g * 4 + 0, 32);
  float v1 = __shfl(sfc, g * 4 + 1, 32);
  float v2 = __shfl(sfc, g * 4 + 2, 32);
  float v3 = __shfl(sfc, g * 4 + 3, 32);
  float a = fmaxf(v0, v1), b = fminf(v0, v1);
  float c = fmaxf(v2, v3), d = fminf(v2, v3);
  float gsc = fmaxf(a, c) + fmaxf(fminf(a, c), fmaxf(b, d));

  float gs[8];
#pragma unroll
  for (int j = 0; j < 8; ++j) gs[j] = __shfl(gsc, j * 4, 32);

  // top-4 groups, lowest-index tie-break (matches lax.top_k)
  unsigned gmask = 0;
#pragma unroll
  for (int k = 0; k < 4; ++k) {
    int best = 0; float bv = -__builtin_inff();
#pragma unroll
    for (int j = 0; j < 8; ++j)
      if (!((gmask >> j) & 1) && gs[j] > bv) { bv = gs[j]; best = j; }
    gmask |= 1u << best;
  }

  float masked = ((gmask >> g) & 1) ? sfc : -__builtin_inff();

  // top-6 experts via iterative wave argmax
  for (int k = 0; k < TOPK; ++k) {
    float mv = masked; int mi = lane;
#pragma unroll
    for (int off = 16; off > 0; off >>= 1) {
      float ov = __shfl_xor(mv, off, 32);
      int   oi = __shfl_xor(mi, off, 32);
      if (ov > mv || (ov == mv && oi < mi)) { mv = ov; mi = oi; }
    }
    float wsel = __shfl(score, mi, 32);   // weight uses raw sigmoid score
    if (lane == k) {
      topk_e[t * TOPK + k] = mi;
      topk_w[t * TOPK + k] = wsel;
      pos_arr[t * TOPK + k] = atomicAdd(&expert_count[mi], 1);
    }
    if (lane == mi) masked = -__builtin_inff();
  }
}

__global__ void scan_kernel(const int* __restrict__ count, int* __restrict__ offsets) {
  if (threadIdx.x == 0 && blockIdx.x == 0) {
    int s = 0;
    for (int e = 0; e < E; ++e) { offsets[e] = s; s += count[e]; }
    offsets[E] = s;
  }
}

__global__ void scatter_kernel(const int* __restrict__ topk_e,
                               const int* __restrict__ pos_arr,
                               const int* __restrict__ offsets,
                               int* __restrict__ tok_list,
                               int* __restrict__ topk_slot)
{
  int s = blockIdx.x * blockDim.x + threadIdx.x;
  if (s >= NSLOT) return;
  int slot = offsets[topk_e[s]] + pos_arr[s];
  tok_list[slot] = s / TOPK;
  topk_slot[s] = slot;
}

// ---------------------------------------------------------------- routed gate/up
// Block = 4 waves stacked over 128 token rows, ONE 16-col m-tile: all waves
// issue identical B addresses -> WGP$ hits, ~4x less L2 weight traffic.
// Each wave: 32 rows (two A fragments), B reused across both.
// grid (ceil(T/128), M/16, E)
__global__ __launch_bounds__(128) void moe_gateup_kernel(
    const float* __restrict__ x, const float* __restrict__ gate_w,
    const float* __restrict__ up_w, const int* __restrict__ expert_count,
    const int* __restrict__ offsets, const int* __restrict__ tok_list,
    _Float16* __restrict__ act_rt)
{
  const int e = blockIdx.z;
  const int cnt = expert_count[e];
  const int wave = threadIdx.x >> 5, lane = threadIdx.x & 31;
  const int row0 = blockIdx.x * 128 + wave * 32;
  if (row0 >= cnt) return;

  const int cb = blockIdx.y * 16;                // m col base
  const int base = offsets[e] + row0;
  const int r = lane & 15, hl = lane >> 4;

  const bool va0 = (row0 + r) < cnt;
  const bool va1 = (row0 + 16 + r) < cnt;
  const int tok0 = va0 ? tok_list[base + r] : tok_list[base];
  const int tok1 = va1 ? tok_list[base + 16 + r] : tok_list[base];
  const float* xr0 = x + (size_t)tok0 * H;
  const float* xr1 = x + (size_t)tok1 * H;
  const float* gw = gate_w + (size_t)e * H * M;
  const float* uw = up_w   + (size_t)e * H * M;

  v8f ag0 = {}, au0 = {}, ag1 = {}, au1 = {};
  for (int k0 = 0; k0 < H; k0 += 32) {
    __builtin_prefetch(&gw[(size_t)(k0 + 128 + 16 * hl) * M + cb + r], 0, 0);
    __builtin_prefetch(&uw[(size_t)(k0 + 128 + 16 * hl) * M + cb + r], 0, 0);
    v16h a0 = load_a_f32(xr0, k0 + 8 * hl);
    v16h a1 = load_a_f32(xr1, k0 + 8 * hl);
    v16h bg = load_b_f32(gw, M, k0 + 8 * hl, cb + r);
    v16h bu = load_b_f32(uw, M, k0 + 8 * hl, cb + r);
    ag0 = wmma16(a0, bg, ag0);
    au0 = wmma16(a0, bu, au0);
    ag1 = wmma16(a1, bg, ag1);
    au1 = wmma16(a1, bu, au1);
  }

  const int n = lane & 15;
#pragma unroll
  for (int vr = 0; vr < 8; ++vr) {
    int m = vr + 8 * hl;
    if (row0 + m < cnt)
      act_rt[(size_t)(base + m) * M + cb + n] = (_Float16)silu_mul(ag0[vr], au0[vr]);
    if (row0 + 16 + m < cnt)
      act_rt[(size_t)(base + 16 + m) * M + cb + n] =
          (_Float16)silu_mul(ag1[vr], au1[vr]);
  }
}

// ---------------------------------------------------------------- routed down
// grid (ceil(T/128), H/16, E), block = 4 waves stacked over rows
__global__ __launch_bounds__(128) void moe_down_kernel(
    const _Float16* __restrict__ act_rt, const float* __restrict__ down_w,
    const int* __restrict__ expert_count, const int* __restrict__ offsets,
    float* __restrict__ down_out)
{
  const int e = blockIdx.z;
  const int cnt = expert_count[e];
  const int wave = threadIdx.x >> 5, lane = threadIdx.x & 31;
  const int row0 = blockIdx.x * 128 + wave * 32;
  if (row0 >= cnt) return;

  const int cb = blockIdx.y * 16;                // h col base
  const int base = offsets[e] + row0;
  const int r = lane & 15, hl = lane >> 4;

  const int rr0 = ((row0 + r) < cnt) ? r : 0;
  const int rr1 = ((row0 + 16 + r) < cnt) ? 16 + r : 0;
  const _Float16* ar0 = act_rt + (size_t)(base + rr0) * M;
  const _Float16* ar1 = act_rt + (size_t)(base + rr1) * M;
  const float* dw = down_w + (size_t)e * M * H;

  v8f acc0 = {}, acc1 = {};
  for (int k0 = 0; k0 < M; k0 += 32) {
    __builtin_prefetch(&dw[(size_t)(k0 + 128 + 16 * hl) * H + cb + r], 0, 0);
    v16h a0 = load_a_f16(ar0, k0 + 8 * hl);
    v16h a1 = load_a_f16(ar1, k0 + 8 * hl);
    v16h bf = load_b_f32(dw, H, k0 + 8 * hl, cb + r);
    acc0 = wmma16(a0, bf, acc0);
    acc1 = wmma16(a1, bf, acc1);
  }

  const int n = lane & 15;
#pragma unroll
  for (int vr = 0; vr < 8; ++vr) {
    int m = vr + 8 * hl;
    if (row0 + m < cnt)
      down_out[(size_t)(base + m) * H + cb + n] = acc0[vr];
    if (row0 + 16 + m < cnt)
      down_out[(size_t)(base + 16 + m) * H + cb + n] = acc1[vr];
  }
}

// ---------------------------------------------------------------- shared expert
// grid (T/128, MSH/16), block = 4 waves stacked over rows
__global__ __launch_bounds__(128) void shared_gateup_kernel(
    const float* __restrict__ x, const float* __restrict__ gw,
    const float* __restrict__ uw, _Float16* __restrict__ act_sh)
{
  const int wave = threadIdx.x >> 5, lane = threadIdx.x & 31;
  const int row0 = blockIdx.x * 128 + wave * 32;
  const int cb = blockIdx.y * 16;
  const int r = lane & 15, hl = lane >> 4;
  const float* xr0 = x + (size_t)(row0 + r) * H;
  const float* xr1 = x + (size_t)(row0 + 16 + r) * H;

  v8f ag0 = {}, au0 = {}, ag1 = {}, au1 = {};
  for (int k0 = 0; k0 < H; k0 += 32) {
    __builtin_prefetch(&gw[(size_t)(k0 + 128 + 16 * hl) * MSH + cb + r], 0, 0);
    __builtin_prefetch(&uw[(size_t)(k0 + 128 + 16 * hl) * MSH + cb + r], 0, 0);
    v16h a0 = load_a_f32(xr0, k0 + 8 * hl);
    v16h a1 = load_a_f32(xr1, k0 + 8 * hl);
    v16h bg = load_b_f32(gw, MSH, k0 + 8 * hl, cb + r);
    v16h bu = load_b_f32(uw, MSH, k0 + 8 * hl, cb + r);
    ag0 = wmma16(a0, bg, ag0);
    au0 = wmma16(a0, bu, au0);
    ag1 = wmma16(a1, bg, ag1);
    au1 = wmma16(a1, bu, au1);
  }

  const int n = lane & 15;
#pragma unroll
  for (int vr = 0; vr < 8; ++vr) {
    int m = vr + 8 * hl;
    act_sh[(size_t)(row0 + m) * MSH + cb + n] =
        (_Float16)silu_mul(ag0[vr], au0[vr]);
    act_sh[(size_t)(row0 + 16 + m) * MSH + cb + n] =
        (_Float16)silu_mul(ag1[vr], au1[vr]);
  }
}

// grid (T/128, H/16), block = 4 waves stacked over rows
__global__ __launch_bounds__(128) void shared_down_kernel(
    const _Float16* __restrict__ act_sh, const float* __restrict__ dw,
    float* __restrict__ out)
{
  const int wave = threadIdx.x >> 5, lane = threadIdx.x & 31;
  const int row0 = blockIdx.x * 128 + wave * 32;
  const int cb = blockIdx.y * 16;
  const int r = lane & 15, hl = lane >> 4;
  const _Float16* ar0 = act_sh + (size_t)(row0 + r) * MSH;
  const _Float16* ar1 = act_sh + (size_t)(row0 + 16 + r) * MSH;

  v8f acc0 = {}, acc1 = {};
  for (int k0 = 0; k0 < MSH; k0 += 32) {
    __builtin_prefetch(&dw[(size_t)(k0 + 128 + 16 * hl) * H + cb + r], 0, 0);
    v16h a0 = load_a_f16(ar0, k0 + 8 * hl);
    v16h a1 = load_a_f16(ar1, k0 + 8 * hl);
    v16h bf = load_b_f32(dw, H, k0 + 8 * hl, cb + r);
    acc0 = wmma16(a0, bf, acc0);
    acc1 = wmma16(a1, bf, acc1);
  }

  const int n = lane & 15;
#pragma unroll
  for (int vr = 0; vr < 8; ++vr) {
    int m = vr + 8 * hl;
    out[(size_t)(row0 + m) * H + cb + n] = acc0[vr];
    out[(size_t)(row0 + 16 + m) * H + cb + n] = acc1[vr];
  }
}

// ---------------------------------------------------------------- final combine
__global__ void combine_kernel(float* __restrict__ out,
                               const float* __restrict__ down_out,
                               const float* __restrict__ topk_w,
                               const int* __restrict__ topk_slot)
{
  int idx = blockIdx.x * blockDim.x + threadIdx.x;
  if (idx >= T * H) return;
  int t = idx / H, h = idx % H;
  float s = out[idx];
#pragma unroll
  for (int k = 0; k < TOPK; ++k) {
    s += topk_w[t * TOPK + k] *
         down_out[(size_t)topk_slot[t * TOPK + k] * H + h];
  }
  out[idx] = s;
}

// ---------------------------------------------------------------- launch
extern "C" void kernel_launch(void* const* d_in, const int* in_sizes, int n_in,
                              void* d_out, int out_size, void* d_ws, size_t ws_size,
                              hipStream_t stream) {
  const float* x         = (const float*)d_in[0];
  const float* w_router  = (const float*)d_in[1];
  const float* corr_bias = (const float*)d_in[2];
  const float* gate_w    = (const float*)d_in[3];
  const float* up_w      = (const float*)d_in[4];
  const float* down_w    = (const float*)d_in[5];
  const float* sh_gate_w = (const float*)d_in[6];
  const float* sh_up_w   = (const float*)d_in[7];
  const float* sh_down_w = (const float*)d_in[8];
  float* out = (float*)d_out;

  // workspace layout (256B aligned slabs)
  char* ws = (char*)d_ws;
  size_t o = 0;
  auto alloc = [&](size_t bytes) { size_t r = o; o = (o + bytes + 255) & ~(size_t)255; return r; };
  int*      expert_count = (int*)     (ws + alloc(E * sizeof(int)));
  int*      offsets      = (int*)     (ws + alloc((E + 1) * sizeof(int)));
  int*      topk_e       = (int*)     (ws + alloc(NSLOT * sizeof(int)));
  float*    topk_w       = (float*)   (ws + alloc(NSLOT * sizeof(float)));
  int*      pos_arr      = (int*)     (ws + alloc(NSLOT * sizeof(int)));
  int*      topk_slot    = (int*)     (ws + alloc(NSLOT * sizeof(int)));
  int*      tok_list     = (int*)     (ws + alloc(NSLOT * sizeof(int)));
  _Float16* act_sh       = (_Float16*)(ws + alloc((size_t)T * MSH * sizeof(_Float16)));
  _Float16* act_rt       = (_Float16*)(ws + alloc((size_t)NSLOT * M * sizeof(_Float16)));
  float*    down_out     = (float*)   (ws + alloc((size_t)NSLOT * H * sizeof(float)));
  (void)ws_size; (void)in_sizes; (void)n_in; (void)out_size;

  hipMemsetAsync(expert_count, 0, E * sizeof(int), stream);

  router_kernel<<<T, 32, 0, stream>>>(x, w_router, corr_bias,
                                      expert_count, topk_e, topk_w, pos_arr);
  scan_kernel<<<1, 32, 0, stream>>>(expert_count, offsets);
  scatter_kernel<<<(NSLOT + 255) / 256, 256, 0, stream>>>(topk_e, pos_arr, offsets,
                                                          tok_list, topk_slot);

  moe_gateup_kernel<<<dim3(T / 128, M / 16, E), 128, 0, stream>>>(
      x, gate_w, up_w, expert_count, offsets, tok_list, act_rt);
  moe_down_kernel<<<dim3(T / 128, H / 16, E), 128, 0, stream>>>(
      act_rt, down_w, expert_count, offsets, down_out);

  shared_gateup_kernel<<<dim3(T / 128, MSH / 16), 128, 0, stream>>>(
      x, sh_gate_w, sh_up_w, act_sh);
  shared_down_kernel<<<dim3(T / 128, H / 16), 128, 0, stream>>>(
      act_sh, sh_down_w, out);

  combine_kernel<<<(T * H + 255) / 256, 256, 0, stream>>>(out, down_out,
                                                          topk_w, topk_slot);
}
